// GCN_31585189495344
// MI455X (gfx1250) — matrix operand order repaired and checked by reading, not compiled, not run on hardware
//
#include <hip/hip_runtime.h>
#include <hip/hip_bf16.h>

// ---------------------------------------------------------------------------
// Types for CDNA5 WMMA (wave32, 16x16x32 bf16 -> f32)
// ---------------------------------------------------------------------------
typedef __attribute__((ext_vector_type(16))) __bf16 v16bf;
typedef __attribute__((ext_vector_type(8)))  __bf16 v8bf;
typedef __attribute__((ext_vector_type(4)))  __bf16 v4bf;
typedef __attribute__((ext_vector_type(8)))  float  v8f;
typedef __attribute__((ext_vector_type(4)))  float  v4f;

__device__ __forceinline__ __bf16 f2bf(float x) { return (__bf16)x; }
__device__ __forceinline__ float  bf2f(__bf16 x) { return (float)x; }

__device__ __forceinline__ v4bf pack4(float a, float b, float c, float d) {
  v4bf r; r[0] = f2bf(a); r[1] = f2bf(b); r[2] = f2bf(c); r[3] = f2bf(d);
  return r;
}

__device__ __forceinline__ float lrelu01(float x) { return x > 0.f ? x : 0.01f * x; }

// ---------------------------------------------------------------------------
// Problem constants
// ---------------------------------------------------------------------------
#define NN 8192
#define D0 1024
#define D1 512
#define D2 256
#define D3 128
#define ALPHA_C 100.0f
#define BETA_C  1.0f
#define ETA_C   25.0f

// GEMM A-operand modes
#define AMODE_F32  0   // A is fp32 in global, convert to bf16 while staging
#define AMODE_BF16 1   // A is bf16 in global (adj_bf16)
#define AMODE_ATTN 2   // A is adj_bf16; compute softmax coef on the fly
// Epilogue modes
#define EPI_NONE         0
#define EPI_BIAS_LRELU   1
#define EPI_LRELU        2
#define EPI_BIAS_SIGMOID 3

// ---------------------------------------------------------------------------
// Tiled WMMA GEMM:  C(MxN) = epilogue( A(MxK) @ B(KxN) [+bias] )
// Block tile 64x64, K-step 32. 128 threads = 4 waves, each wave 32x32
// (2x2 fragments of v_wmma_f32_16x16x32_bf16).
// LDS layouts chosen so every fragment = two aligned ds_load_b128:
//   As  row-major [64][40]: A-frag lane(h,l) = As[row][8h..8h+7] ++ As[row][16+8h..]
//   BsT col-major [64][40]: B-frag lane(h,l) = BsT[col][8h..8h+7] ++ BsT[col][16+8h..]
// Stride 40 bf16 = 80B (16B-aligned rows, 20-dword pitch -> conflict-friendly).
// All M,N multiples of 64 and K multiples of 32 (true for every call here).
// ---------------------------------------------------------------------------
template <int AMODE, int EPI, int TRANSB>
__global__ void __launch_bounds__(128)
gemm_wmma(int M, int N, int K,
          const void* __restrict__ Av, int lda,
          const float* __restrict__ B, int ldb,
          float* __restrict__ C, int ldc,
          const float* __restrict__ bias,
          const float* __restrict__ f1, const float* __restrict__ f2,
          const float* __restrict__ rmax, const float* __restrict__ rsinv) {
  __shared__ __bf16 As[64][40];
  __shared__ __bf16 BsT[64][40];

  const int tid  = threadIdx.x;
  const int lane = tid & 31;
  const int wid  = tid >> 5;
  const int wm   = (wid >> 1) * 32;   // wave row offset in block tile
  const int wn   = (wid & 1) * 32;    // wave col offset
  const int l    = lane & 15;
  const int h    = lane >> 4;

  const int bm0 = blockIdx.y * 64;
  const int bn0 = blockIdx.x * 64;

  const float*  Af = (const float*)Av;
  const __bf16* Ab = (const __bf16*)Av;

  v8f acc[2][2] = {};

  for (int k0 = 0; k0 < K; k0 += 32) {
    // ---- prefetch next K tile (global_prefetch_b8) ----
    if (k0 + 32 < K) {
      if (tid < 64) {
        if (AMODE == AMODE_F32)
          __builtin_prefetch(Af + (size_t)(bm0 + tid) * lda + k0 + 32, 0, 3);
        else
          __builtin_prefetch(Ab + (size_t)(bm0 + tid) * lda + k0 + 32, 0, 3);
      } else if (tid < 96 && !TRANSB) {
        __builtin_prefetch(B + (size_t)(k0 + 32 + (tid - 64)) * ldb + bn0, 0, 3);
      }
    }

    // ---- stage A tile: 64x32 (packed v4bf -> single 8B LDS store) ----
    #pragma unroll
    for (int e = tid * 4; e < 64 * 32; e += 128 * 4) {
      const int r = e >> 5, c = e & 31;
      const int gm = bm0 + r, gk = k0 + c;
      if (AMODE == AMODE_F32) {
        v4f v = *reinterpret_cast<const v4f*>(Af + (size_t)gm * lda + gk);
        *reinterpret_cast<v4bf*>(&As[r][c]) = pack4(v[0], v[1], v[2], v[3]);
      } else if (AMODE == AMODE_BF16) {
        *reinterpret_cast<v4bf*>(&As[r][c]) =
            *reinterpret_cast<const v4bf*>(Ab + (size_t)gm * lda + gk);
      } else { // AMODE_ATTN: coefs[i][j] = exp(masked_lrelu(f1+f2) - rowmax) * rowsuminv
        v4bf v = *reinterpret_cast<const v4bf*>(Ab + (size_t)gm * lda + gk);
        const float fi = f1[gm], mi = rmax[gm], si = rsinv[gm];
        v4bf o;
        #pragma unroll
        for (int q = 0; q < 4; ++q) {
          float a   = bf2f(v[q]);
          float x   = fi + f2[gk + q];
          float val = lrelu01(x);
          if (a <= 0.f) val -= 1.0e9f;
          o[q] = f2bf(__expf(val - mi) * si);
        }
        *reinterpret_cast<v4bf*>(&As[r][c]) = o;
      }
    }

    // ---- stage B tile transposed into BsT[n][k] ----
    if (!TRANSB) {
      // 4 coalesced b32 loads (consecutive n across lanes), one b64 LDS store
      #pragma unroll
      for (int e = tid; e < 64 * 8; e += 128) {
        const int n  = e & 63;
        const int kb = (e >> 6) * 4;
        const float* bp = B + (size_t)(k0 + kb) * ldb + bn0 + n;
        *reinterpret_cast<v4bf*>(&BsT[n][kb]) =
            pack4(bp[0], bp[(size_t)ldb], bp[2 * (size_t)ldb], bp[3 * (size_t)ldb]);
      }
    } else { // B logical[k][n] = Bmem[n*ldb + k]  (attn_W.T): contiguous along k
      #pragma unroll
      for (int e = tid; e < 64 * 8; e += 128) {
        const int n  = e >> 3;
        const int kb = (e & 7) * 4;
        v4f v = *reinterpret_cast<const v4f*>(B + (size_t)(bn0 + n) * ldb + k0 + kb);
        *reinterpret_cast<v4bf*>(&BsT[n][kb]) = pack4(v[0], v[1], v[2], v[3]);
      }
    }
    __syncthreads();

    // ---- fragments: two ds_load_b128 each ----
    v16bf afr[2], bfr[2];
    #pragma unroll
    for (int i = 0; i < 2; ++i) {
      const int row = wm + i * 16 + l;
      v8bf lo = *reinterpret_cast<const v8bf*>(&As[row][8 * h]);
      v8bf hi = *reinterpret_cast<const v8bf*>(&As[row][16 + 8 * h]);
      afr[i] = __builtin_shufflevector(lo, hi, 0, 1, 2, 3, 4, 5, 6, 7,
                                       8, 9, 10, 11, 12, 13, 14, 15);
    }
    #pragma unroll
    for (int j = 0; j < 2; ++j) {
      const int col = wn + j * 16 + l;
      v8bf lo = *reinterpret_cast<const v8bf*>(&BsT[col][8 * h]);
      v8bf hi = *reinterpret_cast<const v8bf*>(&BsT[col][16 + 8 * h]);
      bfr[j] = __builtin_shufflevector(lo, hi, 0, 1, 2, 3, 4, 5, 6, 7,
                                       8, 9, 10, 11, 12, 13, 14, 15);
    }

    #pragma unroll
    for (int i = 0; i < 2; ++i)
      #pragma unroll
      for (int j = 0; j < 2; ++j)
        acc[i][j] = __builtin_amdgcn_wmma_f32_16x16x32_bf16(
            false, afr[i], false, bfr[j], (short)0, acc[i][j], false, false);

    __syncthreads();
  }

  // ---- epilogue + store (C layout: VGPR q -> M = q + 8*h, N = l) ----
  #pragma unroll
  for (int i = 0; i < 2; ++i) {
    #pragma unroll
    for (int j = 0; j < 2; ++j) {
      const int n = bn0 + wn + j * 16 + l;
      const float bv = (EPI == EPI_BIAS_LRELU || EPI == EPI_BIAS_SIGMOID) ? bias[n] : 0.f;
      #pragma unroll
      for (int q = 0; q < 8; ++q) {
        const int m = bm0 + wm + i * 16 + q + 8 * h;
        float v = acc[i][j][q] + bv;
        if (EPI == EPI_BIAS_LRELU || EPI == EPI_LRELU) v = lrelu01(v);
        if (EPI == EPI_BIAS_SIGMOID) v = 1.f / (1.f + __expf(-v));
        C[(size_t)m * ldc + n] = v;
      }
    }
  }
}

// ---------------------------------------------------------------------------
// adj(fp32) -> adj(bf16) conversion fused with degree rowsum. One wave per row.
// ---------------------------------------------------------------------------
__global__ void __launch_bounds__(256)
conv_deg_kernel(const float* __restrict__ adj, __bf16* __restrict__ adjbf,
                float* __restrict__ deg) {
  const int row  = (blockIdx.x * blockDim.x + threadIdx.x) >> 5;
  const int lane = threadIdx.x & 31;
  if (row >= NN) return;
  const float* ar = adj + (size_t)row * NN;
  __bf16* br = adjbf + (size_t)row * NN;
  float s = 0.f;
  for (int c = lane * 4; c < NN; c += 32 * 4) {
    v4f v = *reinterpret_cast<const v4f*>(ar + c);
    s += v[0] + v[1] + v[2] + v[3];
    *reinterpret_cast<v4bf*>(br + c) = pack4(v[0], v[1], v[2], v[3]);
  }
  #pragma unroll
  for (int off = 16; off; off >>= 1) s += __shfl_xor(s, off, 32);
  if (lane == 0) deg[row] = s;
}

// ---------------------------------------------------------------------------
// f1 = seq_fts @ a1, f2 = seq_fts @ a2   (one wave per row, D2=256 cols)
// ---------------------------------------------------------------------------
__global__ void __launch_bounds__(256)
matvec2_kernel(const float* __restrict__ S, const float* __restrict__ a1,
               const float* __restrict__ a2, float* __restrict__ f1,
               float* __restrict__ f2) {
  const int row  = (blockIdx.x * blockDim.x + threadIdx.x) >> 5;
  const int lane = threadIdx.x & 31;
  if (row >= NN) return;
  const float* r = S + (size_t)row * D2;
  float s1 = 0.f, s2 = 0.f;
  for (int c = lane; c < D2; c += 32) {
    float v = r[c];
    s1 += v * a1[c];
    s2 += v * a2[c];
  }
  #pragma unroll
  for (int off = 16; off; off >>= 1) {
    s1 += __shfl_xor(s1, off, 32);
    s2 += __shfl_xor(s2, off, 32);
  }
  if (lane == 0) { f1[row] = s1; f2[row] = s2; }
}

// ---------------------------------------------------------------------------
// Online softmax row stats over masked lrelu logits. One block (256t) per row.
// ---------------------------------------------------------------------------
__global__ void __launch_bounds__(256)
attn_stats_kernel(const __bf16* __restrict__ adjbf, const float* __restrict__ f1,
                  const float* __restrict__ f2, float* __restrict__ rmax,
                  float* __restrict__ rsinv) {
  const int row = blockIdx.x;
  const int tid = threadIdx.x;
  const float fi = f1[row];
  float m = -3.0e38f, s = 0.f;
  for (int j = tid; j < NN; j += 256) {
    float a   = bf2f(adjbf[(size_t)row * NN + j]);
    float x   = fi + f2[j];
    float val = lrelu01(x);
    if (a <= 0.f) val -= 1.0e9f;
    if (val > m) { s = s * __expf(m - val) + 1.f; m = val; }
    else          { s += __expf(val - m); }
  }
  __shared__ float sm[256], ss[256];
  sm[tid] = m; ss[tid] = s;
  __syncthreads();
  for (int st = 128; st; st >>= 1) {
    if (tid < st) {
      float m2 = sm[tid + st], s2 = ss[tid + st];
      float M = fmaxf(sm[tid], m2);
      ss[tid] = ss[tid] * __expf(sm[tid] - M) + s2 * __expf(m2 - M);
      sm[tid] = M;
    }
    __syncthreads();
  }
  if (tid == 0) { rmax[row] = sm[0]; rsinv[row] = 1.f / ss[0]; }
}

// ---------------------------------------------------------------------------
// Reductions for the two loss terms
// ---------------------------------------------------------------------------
__device__ __forceinline__ float block_sum256(float v) {
  __shared__ float sm[256];
  const int t = threadIdx.x;
  sm[t] = v;
  __syncthreads();
  for (int s = 128; s; s >>= 1) {
    if (t < s) sm[t] += sm[t + s];
    __syncthreads();
  }
  return sm[0];
}

__global__ void __launch_bounds__(256)
zero_acc_kernel(float* acc) { acc[0] = 0.f; acc[1] = 0.f; }

// loss_emb partial: sum_i H[i]*(deg[row]*H[i] - adjH[i]), total = NN*D3
__global__ void __launch_bounds__(256)
emb_reduce_kernel(const float* __restrict__ H, const float* __restrict__ deg,
                  const float* __restrict__ adjH, float* __restrict__ acc) {
  const size_t total = (size_t)NN * D3;
  float s = 0.f;
  for (size_t i = blockIdx.x * 256 + threadIdx.x; i < total;
       i += (size_t)gridDim.x * 256) {
    const int row = (int)(i >> 7);   // D3 == 128
    const float t = H[i];
    s += t * (deg[row] * t - adjH[i]);
  }
  s = block_sum256(s);
  if (threadIdx.x == 0) atomicAdd(acc + 1, s);
}

// loss_res partial: sum ((Xr - X) * (X*(ETA-1)+1))^2, total = NN*D0
__global__ void __launch_bounds__(256)
res_reduce_kernel(const float* __restrict__ Xr, const float* __restrict__ X,
                  float* __restrict__ acc) {
  const size_t total = (size_t)NN * D0;
  float s = 0.f;
  for (size_t i = blockIdx.x * 256 + threadIdx.x; i < total;
       i += (size_t)gridDim.x * 256) {
    const float x = X[i];
    const float b = x * (ETA_C - 1.f) + 1.f;
    const float d = (Xr[i] - x) * b;
    s += d * d;
  }
  s = block_sum256(s);
  if (threadIdx.x == 0) atomicAdd(acc + 0, s);
}

__global__ void __launch_bounds__(256)
final_loss_kernel(const float* __restrict__ acc, float* __restrict__ out) {
  out[0] = ALPHA_C * acc[0] + BETA_C * (2.f * acc[1]);
}

__global__ void __launch_bounds__(256)
copy_kernel(const float* __restrict__ src, float* __restrict__ dst, size_t n) {
  for (size_t i = blockIdx.x * 256 + threadIdx.x; i < n;
       i += (size_t)gridDim.x * 256)
    dst[i] = src[i];
}

// ---------------------------------------------------------------------------
// Host launcher
// ---------------------------------------------------------------------------
extern "C" void kernel_launch(void* const* d_in, const int* in_sizes, int n_in,
                              void* d_out, int out_size, void* d_ws, size_t ws_size,
                              hipStream_t stream) {
  (void)in_sizes; (void)n_in; (void)out_size; (void)ws_size;
  const float* X      = (const float*)d_in[0];
  const float* adj    = (const float*)d_in[1];
  const float* W0     = (const float*)d_in[2];
  const float* b0     = (const float*)d_in[3];
  const float* W1     = (const float*)d_in[4];
  const float* b1     = (const float*)d_in[5];
  const float* W2     = (const float*)d_in[6];
  const float* b2     = (const float*)d_in[7];
  const float* attnW  = (const float*)d_in[8];
  const float* a1     = (const float*)d_in[9];
  const float* a2     = (const float*)d_in[10];
  const float* dW0    = (const float*)d_in[11];
  const float* db0    = (const float*)d_in[12];
  const float* dW1    = (const float*)d_in[13];
  const float* db1    = (const float*)d_in[14];
  const float* dW2    = (const float*)d_in[15];
  const float* db2    = (const float*)d_in[16];

  float* out = (float*)d_out;
  float* outH  = out + 1;
  float* outXr = out + 1 + (size_t)NN * D3;

  // ---- workspace layout (~228 MB) ----
  char* ws = (char*)d_ws;
  const size_t SNN = (size_t)NN;
  size_t o = 0;
  __bf16* adjbf = (__bf16*)(ws + o); o += SNN * SNN * 2;          // 128 MB
  float* XW0    = (float*)(ws + o);  o += SNN * D1 * 4;
  float* h0     = (float*)(ws + o);  o += SNN * D1 * 4;
  float* h0W1   = (float*)(ws + o);  o += SNN * D2 * 4;
  float* h1     = (float*)(ws + o);  o += SNN * D2 * 4;
  float* seq    = (float*)(ws + o);  o += SNN * D2 * 4;
  float* h2     = (float*)(ws + o);  o += SNN * D2 * 4;
  float* h2W2   = (float*)(ws + o);  o += SNN * D3 * 4;
  float* Hws    = (float*)(ws + o);  o += SNN * D3 * 4;
  float* adjH   = (float*)(ws + o);  o += SNN * D3 * 4;
  float* Xr0    = (float*)(ws + o);  o += SNN * D2 * 4;
  float* Xr1    = (float*)(ws + o);  o += SNN * D1 * 4;
  float* f1     = (float*)(ws + o);  o += SNN * 4;
  float* f2     = (float*)(ws + o);  o += SNN * 4;
  float* rmax   = (float*)(ws + o);  o += SNN * 4;
  float* rsinv  = (float*)(ws + o);  o += SNN * 4;
  float* deg    = (float*)(ws + o);  o += SNN * 4;
  float* acc    = (float*)(ws + o);  o += 256;

  const dim3 gblk(128);
  auto ggrid = [](int M, int N) { return dim3(N / 64, M / 64); };

  // 0. zero loss accumulators (independent)
  zero_acc_kernel<<<1, 256, 0, stream>>>(acc);

  // 1. adj -> bf16 + degree (one wave per row; 8192 waves)
  conv_deg_kernel<<<1024, 256, 0, stream>>>(adj, adjbf, deg);

  // 2. XW0 = X @ W0
  gemm_wmma<AMODE_F32, EPI_NONE, 0><<<ggrid(NN, D1), gblk, 0, stream>>>(
      NN, D1, D0, X, D0, W0, D1, XW0, D1, nullptr, nullptr, nullptr, nullptr, nullptr);
  // 3. h0 = lrelu(adj @ XW0 + b0)
  gemm_wmma<AMODE_BF16, EPI_BIAS_LRELU, 0><<<ggrid(NN, D1), gblk, 0, stream>>>(
      NN, D1, NN, adjbf, NN, XW0, D1, h0, D1, b0, nullptr, nullptr, nullptr, nullptr);
  // 4. h0W1 = h0 @ W1
  gemm_wmma<AMODE_F32, EPI_NONE, 0><<<ggrid(NN, D2), gblk, 0, stream>>>(
      NN, D2, D1, h0, D1, W1, D2, h0W1, D2, nullptr, nullptr, nullptr, nullptr, nullptr);
  // 5. h1 = lrelu(adj @ h0W1 + b1)
  gemm_wmma<AMODE_BF16, EPI_BIAS_LRELU, 0><<<ggrid(NN, D2), gblk, 0, stream>>>(
      NN, D2, NN, adjbf, NN, h0W1, D2, h1, D2, b1, nullptr, nullptr, nullptr, nullptr);
  // 6. seq_fts = h1 @ attn_W.T
  gemm_wmma<AMODE_F32, EPI_NONE, 1><<<ggrid(NN, D2), gblk, 0, stream>>>(
      NN, D2, D2, h1, D2, attnW, D2, seq, D2, nullptr, nullptr, nullptr, nullptr, nullptr);
  // 7. f1/f2 matvecs
  matvec2_kernel<<<1024, 256, 0, stream>>>(seq, a1, a2, f1, f2);
  // 8. softmax row stats (online max/sum)
  attn_stats_kernel<<<NN, 256, 0, stream>>>(adjbf, f1, f2, rmax, rsinv);
  // 9. h2 = lrelu(coefs @ seq_fts)  — coefs reconstructed in A-staging
  gemm_wmma<AMODE_ATTN, EPI_LRELU, 0><<<ggrid(NN, D2), gblk, 0, stream>>>(
      NN, D2, NN, adjbf, NN, seq, D2, h2, D2, nullptr, f1, f2, rmax, rsinv);
  // 10. h2W2 = h2 @ W2
  gemm_wmma<AMODE_F32, EPI_NONE, 0><<<ggrid(NN, D3), gblk, 0, stream>>>(
      NN, D3, D2, h2, D2, W2, D3, h2W2, D3, nullptr, nullptr, nullptr, nullptr, nullptr);
  // 11. H = lrelu(adj @ h2W2 + b2)
  gemm_wmma<AMODE_BF16, EPI_BIAS_LRELU, 0><<<ggrid(NN, D3), gblk, 0, stream>>>(
      NN, D3, NN, adjbf, NN, h2W2, D3, Hws, D3, b2, nullptr, nullptr, nullptr, nullptr);
  // 12. H -> output slot (outH is misaligned for vector loads, keep Hws for reuse)
  copy_kernel<<<2048, 256, 0, stream>>>(Hws, outH, SNN * D3);
  // 13. adjH = adj @ H
  gemm_wmma<AMODE_BF16, EPI_NONE, 0><<<ggrid(NN, D3), gblk, 0, stream>>>(
      NN, D3, NN, adjbf, NN, Hws, D3, adjH, D3, nullptr, nullptr, nullptr, nullptr, nullptr);
  // 14. Xr0 = sigmoid(H @ dW0 + db0)
  gemm_wmma<AMODE_F32, EPI_BIAS_SIGMOID, 0><<<ggrid(NN, D2), gblk, 0, stream>>>(
      NN, D2, D3, Hws, D3, dW0, D2, Xr0, D2, db0, nullptr, nullptr, nullptr, nullptr);
  // 15. Xr1 = sigmoid(Xr0 @ dW1 + db1)
  gemm_wmma<AMODE_F32, EPI_BIAS_SIGMOID, 0><<<ggrid(NN, D1), gblk, 0, stream>>>(
      NN, D1, D2, Xr0, D2, dW1, D1, Xr1, D1, db1, nullptr, nullptr, nullptr, nullptr);
  // 16. Xr = sigmoid(Xr1 @ dW2 + db2) -> output (scalar stores, alignment OK)
  gemm_wmma<AMODE_F32, EPI_BIAS_SIGMOID, 0><<<ggrid(NN, D0), gblk, 0, stream>>>(
      NN, D0, D1, Xr1, D1, dW2, D0, outXr, D0, db2, nullptr, nullptr, nullptr, nullptr);

  // 17. loss terms
  emb_reduce_kernel<<<1024, 256, 0, stream>>>(Hws, deg, adjH, acc);
  res_reduce_kernel<<<2048, 256, 0, stream>>>(outXr, X, acc);
  final_loss_kernel<<<1, 256, 0, stream>>>(acc, out);
}
